// RingPolicyModule_6399501271527
// MI455X (gfx1250) — compile-verified, matrix operand order checked by reading, not run on hardware
//
#include <hip/hip_runtime.h>

typedef __attribute__((ext_vector_type(16))) _Float16 v16h;
typedef __attribute__((ext_vector_type(8)))  _Float16 v8h;
typedef __attribute__((ext_vector_type(4)))  _Float16 v4h;
typedef __attribute__((ext_vector_type(8)))  float    v8f;

#define H64 64
#define NN  128        // nodes per graph
#define EPG 512        // edges per graph
#define FS  72         // padded LDS row stride (elements); 72*2B = 144B -> 16B aligned rows
#define OUT_PER_B 16257
#define NUPPER 8128    // 128*127/2
#define WSW_N 4096     // 8 tiles * 32 lanes * 16 halves (swizzled 64x64 weight)

// A-matrix fragment (16x32 f16, row-major LDS source). Per ISA 7.12.2:
// lane<16: row M=lane, K in {0..7, 16..23}; lane>=16: same rows, K in {8..15, 24..31}.
// Elements 0..7 and 8..15 are contiguous 16B chunks -> two ds_load_b128.
__device__ __forceinline__ v16h frag_a_rowmajor(const _Float16* base, int ld, int lane) {
  const int half = lane >> 4;
  const int r    = lane & 15;
  const _Float16* p = base + r * ld + half * 8;
  v8h lo = *(const v8h*)(p);        // K offsets 0..7
  v8h hi = *(const v8h*)(p + 16);   // K offsets 16..23
  return __builtin_shufflevector(lo, hi, 0, 1, 2, 3, 4, 5, 6, 7,
                                         8, 9, 10, 11, 12, 13, 14, 15);
}

// B fragment from pre-swizzled weight buffer: one contiguous 32B v16h per lane.
__device__ __forceinline__ v16h frag_b_swizzled(const _Float16* wsw, int nt, int kt, int lane) {
  return *(const v16h*)(wsw + (((nt << 1) + kt) * 32 + lane) * 16);
}

// C = A(128x64 f16) @ B(64x64 f16 pre-swizzled), f32 accumulate, all in LDS.
// 8 waves: wave w owns M-tile w (A fragments hoisted); loops 4 N-tiles; K = 2 wmma k-steps.
__device__ __forceinline__ void gemm_f16(const _Float16* A, const _Float16* wsw, float* C,
                                         int tid) {
  const int wid = tid >> 5, lane = tid & 31;
  const int mt = wid;
  const int half = lane >> 4, n = lane & 15;
  const v16h a0 = frag_a_rowmajor(A + mt * 16 * FS + 0,  FS, lane);
  const v16h a1 = frag_a_rowmajor(A + mt * 16 * FS + 32, FS, lane);
#pragma unroll
  for (int nt = 0; nt < 4; ++nt) {
    v8f c = {};
    c = __builtin_amdgcn_wmma_f32_16x16x32_f16(false, a0, false, frag_b_swizzled(wsw, nt, 0, lane),
                                               (short)0, c, false, false);
    c = __builtin_amdgcn_wmma_f32_16x16x32_f16(false, a1, false, frag_b_swizzled(wsw, nt, 1, lane),
                                               (short)0, c, false, false);
#pragma unroll
    for (int r = 0; r < 8; ++r)
      C[(mt * 16 + half * 8 + r) * FS + nt * 16 + n] = c[r];
  }
}

// Stage a 64x64 f32 row-major (KxN) weight from global into LDS in B-fragment order:
// wsw[(t*32 + lane)*16 + e] = W[k][n], t = nt*2+kt,
//   k = kt*32 + (lane>>4)*8 + e + (e>=8 ? 8 : 0),  n = nt*16 + (lane&15).
__device__ __forceinline__ void stage_weight(const float* __restrict__ W, _Float16* wsw, int tid) {
  for (int idx = tid; idx < WSW_N; idx += 256) {
    const int e    = idx & 15;
    const int lane = (idx >> 4) & 31;
    const int t    = idx >> 9;
    const int nt   = t >> 1, kt = t & 1;
    const int k = kt * 32 + ((lane >> 4) << 3) + e + ((e >= 8) ? 8 : 0);
    const int n = (nt << 4) + (lane & 15);
    wsw[idx] = (_Float16)W[k * H64 + n];
  }
}

__global__ __launch_bounds__(256)
void ring_policy_kernel(const int* __restrict__ node_feature,
                        const int* __restrict__ edge_index,
                        const float* __restrict__ emb,
                        const float* __restrict__ gcn_w,  const float* __restrict__ gcn_b,
                        const float* __restrict__ lin1_w, const float* __restrict__ lin1_b,
                        const float* __restrict__ lin2_w, const float* __restrict__ lin2_b,
                        const float* __restrict__ ln_g,   const float* __restrict__ ln_b,
                        const float* __restrict__ ex_w1,  const float* __restrict__ ex_b1,
                        const float* __restrict__ ex_ln_g,const float* __restrict__ ex_ln_b,
                        const float* __restrict__ ex_w2,  const float* __restrict__ ex_b2,
                        float* __restrict__ out) {
  extern __shared__ char smem[];
  float*    xs   = (float*)smem;                 // 128*FS f32 : x (kept for residual)
  float*    hs   = xs + NN * FS;                 // 128*FS f32 : gemm results / ef
  float*    aggs = hs + NN * FS;                 // 128*FS f32 : scatter accumulator
  _Float16* ah   = (_Float16*)(aggs + NN * FS);  // 128*FS f16 : A operand / ef f16
  _Float16* wsw  = ah + NN * FS;                 // 4096 f16   : swizzled weight staging
  float*    dinv = (float*)(wsw + WSW_N);        // 128
  float*    degs = dinv + NN;                    // 128
  float*    means= degs + NN;                    // 64
  float*    zbuf = means + H64;                  // 64
  float*    red  = zbuf + H64;                   // 64
  float*    stat = red + H64;                    // 4

  const int b    = blockIdx.x;
  const int tid  = threadIdx.x;
  const int lane = tid & 31, wid = tid >> 5;
  const int2* ei2 = (const int2*)edge_index;

  // Prefetch this block's edge list (4KB) and node-id slice into cache
  // (lowers to global_prefetch_b8, per-lane addresses, no counter cost).
  __builtin_prefetch(((const char*)&ei2[b * EPG]) + tid * 16, 0, 3);
  __builtin_prefetch((const char*)&node_feature[b * NN + (tid & 127)], 0, 3);

  // ---- Phase 1: gather x = emb[node_feature] (float4), init degrees ----
  for (int i = tid; i < NN; i += 256) degs[i] = 1.0f;
  const float4* emb4 = (const float4*)emb;
  for (int idx = tid; idx < NN * 16; idx += 256) {
    const int i = idx >> 4, c = idx & 15;
    const int nid = node_feature[b * NN + i];
    const float4 v = emb4[nid * 16 + c];
    *(float4*)(xs + i * FS + c * 4) = v;
    v4h h; h[0] = (_Float16)v.x; h[1] = (_Float16)v.y; h[2] = (_Float16)v.z; h[3] = (_Float16)v.w;
    *(v4h*)(ah + i * FS + c * 4) = h;
  }
  __syncthreads();
  for (int e = tid; e < EPG; e += 256) {
    const int dl = ei2[b * EPG + e].y - b * NN;
    atomicAdd(&degs[dl], 1.0f);
  }
  stage_weight(gcn_w, wsw, tid);
  __syncthreads();
  for (int i = tid; i < NN; i += 256) dinv[i] = rsqrtf(degs[i]);
  __syncthreads();

  // ---- GEMM1: hs = x @ gcn_w ----
  gemm_f16(ah, wsw, hs, tid);
  __syncthreads();

  // ---- GCN aggregation: aggs = selfloop + gcn_b, then edge scatter-add ----
  for (int idx = tid; idx < NN * 16; idx += 256) {
    const int i = idx >> 4, c = idx & 15;
    const float d2 = dinv[i] * dinv[i];
    const float4 h4 = *(const float4*)(hs + i * FS + c * 4);
    const float4 b4 = *(const float4*)(gcn_b + c * 4);
    float4 o;
    o.x = d2 * h4.x + b4.x; o.y = d2 * h4.y + b4.y;
    o.z = d2 * h4.z + b4.z; o.w = d2 * h4.w + b4.w;
    *(float4*)(aggs + i * FS + c * 4) = o;
  }
  __syncthreads();
  for (int t = tid; t < EPG * 8; t += 256) {
    const int e = t >> 3, fc = (t & 7) * 8;
    const int2 se = ei2[b * EPG + e];
    const int sl = se.x - b * NN, dl = se.y - b * NN;
    const float nrm = dinv[sl] * dinv[dl];
    const float4 a0 = *(const float4*)(hs + sl * FS + fc);
    const float4 a1 = *(const float4*)(hs + sl * FS + fc + 4);
    float* dst = aggs + dl * FS + fc;
    atomicAdd(dst + 0, nrm * a0.x); atomicAdd(dst + 1, nrm * a0.y);
    atomicAdd(dst + 2, nrm * a0.z); atomicAdd(dst + 3, nrm * a0.w);
    atomicAdd(dst + 4, nrm * a1.x); atomicAdd(dst + 5, nrm * a1.y);
    atomicAdd(dst + 6, nrm * a1.z); atomicAdd(dst + 7, nrm * a1.w);
  }
  __syncthreads();

  // ---- GEMM2 prep: ah = f16(aggs), wsw = swizzle(lin1_w) ----
  for (int idx = tid; idx < NN * 16; idx += 256) {
    const int i = idx >> 4, c = idx & 15;
    const float4 v = *(const float4*)(aggs + i * FS + c * 4);
    v4h h; h[0] = (_Float16)v.x; h[1] = (_Float16)v.y; h[2] = (_Float16)v.z; h[3] = (_Float16)v.w;
    *(v4h*)(ah + i * FS + c * 4) = h;
  }
  stage_weight(lin1_w, wsw, tid);
  __syncthreads();
  gemm_f16(ah, wsw, hs, tid);
  __syncthreads();
  // relu(+lin1_b) -> hs (f32) and ah (f16)
  for (int idx = tid; idx < NN * 16; idx += 256) {
    const int i = idx >> 4, c = idx & 15;
    const float4 v = *(const float4*)(hs + i * FS + c * 4);
    const float4 b4 = *(const float4*)(lin1_b + c * 4);
    float4 o;
    o.x = fmaxf(v.x + b4.x, 0.0f); o.y = fmaxf(v.y + b4.y, 0.0f);
    o.z = fmaxf(v.z + b4.z, 0.0f); o.w = fmaxf(v.w + b4.w, 0.0f);
    *(float4*)(hs + i * FS + c * 4) = o;
    v4h h; h[0] = (_Float16)o.x; h[1] = (_Float16)o.y; h[2] = (_Float16)o.z; h[3] = (_Float16)o.w;
    *(v4h*)(ah + i * FS + c * 4) = h;
  }
  stage_weight(lin2_w, wsw, tid);
  __syncthreads();

  // ---- GEMM3: aggs = t @ lin2_w ----
  gemm_f16(ah, wsw, aggs, tid);
  __syncthreads();

  // ---- LayerNorm rows: ef = LN(aggs + lin2_b + x); hs = ef f32, ah = ef f16 ----
  for (int row = wid; row < NN; row += 8) {
    const int f0 = lane, f1 = lane + 32;
    const float v0 = aggs[row * FS + f0] + lin2_b[f0] + xs[row * FS + f0];
    const float v1 = aggs[row * FS + f1] + lin2_b[f1] + xs[row * FS + f1];
    float s = v0 + v1;
#pragma unroll
    for (int off = 16; off > 0; off >>= 1) s += __shfl_xor(s, off, 32);
    const float m = s * (1.0f / 64.0f);
    const float d0 = v0 - m, d1 = v1 - m;
    float q = d0 * d0 + d1 * d1;
#pragma unroll
    for (int off = 16; off > 0; off >>= 1) q += __shfl_xor(q, off, 32);
    const float inv = rsqrtf(q * (1.0f / 64.0f) + 1e-5f);
    const float e0 = d0 * inv * ln_g[f0] + ln_b[f0];
    const float e1 = d1 * inv * ln_g[f1] + ln_b[f1];
    hs[row * FS + f0] = e0;  hs[row * FS + f1] = e1;
    ah[row * FS + f0] = (_Float16)e0;  ah[row * FS + f1] = (_Float16)e1;
  }
  __syncthreads();

  // ---- Exit head: mean over nodes -> linear -> LN -> relu -> dot w2 ----
  if (tid < H64) {
    float s = 0.0f;
    for (int r = 0; r < NN; ++r) s += hs[r * FS + tid];
    means[tid] = s * (1.0f / 128.0f);
  }
  __syncthreads();
  if (tid < H64) {
    float s = ex_b1[tid];
    for (int k = 0; k < H64; ++k) s += means[k] * ex_w1[k * H64 + tid];
    zbuf[tid] = s;
  }
  __syncthreads();
  if (tid == 0) {
    float s = 0.0f;
    for (int k = 0; k < H64; ++k) s += zbuf[k];
    const float m = s * (1.0f / 64.0f);
    float q = 0.0f;
    for (int k = 0; k < H64; ++k) { const float d = zbuf[k] - m; q += d * d; }
    stat[0] = m; stat[1] = rsqrtf(q * (1.0f / 64.0f) + 1e-5f);
  }
  __syncthreads();
  if (tid < H64) {
    float v = (zbuf[tid] - stat[0]) * stat[1] * ex_ln_g[tid] + ex_ln_b[tid];
    red[tid] = fmaxf(v, 0.0f) * ex_w2[tid];
  }
  __syncthreads();
  if (tid == 0) {
    float s = ex_b2[0];
    for (int k = 0; k < H64; ++k) s += red[k];
    out[(size_t)b * OUT_PER_B + 16256] = s;
  }
  __syncthreads();

  // ---- Scores: D = ef @ ef^T * 0.125, permuted triu/tril writeout ----
  {
    const int mt = wid;
    const int halfl = lane >> 4, nl = lane & 15;
    float* outb = out + (size_t)b * OUT_PER_B;
    const v16h a0 = frag_a_rowmajor(ah + mt * 16 * FS + 0,  FS, lane);
    const v16h a1 = frag_a_rowmajor(ah + mt * 16 * FS + 32, FS, lane);
#pragma unroll
    for (int nt = 0; nt < 8; ++nt) {
      v8f c = {};
      v16h b0 = frag_a_rowmajor(ah + nt * 16 * FS + 0,  FS, lane);  // B^T = ef
      v16h b1 = frag_a_rowmajor(ah + nt * 16 * FS + 32, FS, lane);
      c = __builtin_amdgcn_wmma_f32_16x16x32_f16(false, a0, false, b0, (short)0, c, false, false);
      c = __builtin_amdgcn_wmma_f32_16x16x32_f16(false, a1, false, b1, (short)0, c, false, false);
#pragma unroll
      for (int r = 0; r < 8; ++r) {
        const int m = mt * 16 + halfl * 8 + r;
        const int n = nt * 16 + nl;
        if (m == n) continue;
        int idx;
        if (n > m) idx = m * 127 - (m * (m - 1)) / 2 + (n - m - 1);
        else       idx = NUPPER + (m * (m - 1)) / 2 + n;
        outb[idx] = c[r] * 0.125f;
      }
    }
  }
}

extern "C" void kernel_launch(void* const* d_in, const int* in_sizes, int n_in,
                              void* d_out, int out_size, void* d_ws, size_t ws_size,
                              hipStream_t stream) {
  (void)in_sizes; (void)n_in; (void)out_size; (void)d_ws; (void)ws_size;
  const int*   node_feature = (const int*)d_in[0];
  const int*   edge_index   = (const int*)d_in[1];
  // d_in[2] = batch_ptr (uniform blocks of 128 -> unused)
  const float* emb    = (const float*)d_in[3];
  const float* gcn_w  = (const float*)d_in[4];
  const float* gcn_b  = (const float*)d_in[5];
  const float* lin1_w = (const float*)d_in[6];
  const float* lin1_b = (const float*)d_in[7];
  const float* lin2_w = (const float*)d_in[8];
  const float* lin2_b = (const float*)d_in[9];
  const float* ln_g   = (const float*)d_in[10];
  const float* ln_b   = (const float*)d_in[11];
  const float* ex_w1  = (const float*)d_in[12];
  const float* ex_b1  = (const float*)d_in[13];
  const float* ex_ln_g= (const float*)d_in[14];
  const float* ex_ln_b= (const float*)d_in[15];
  const float* ex_w2  = (const float*)d_in[16];
  const float* ex_b2  = (const float*)d_in[17];
  float* out = (float*)d_out;

  const size_t shmem =
      (size_t)(3 * NN * FS) * sizeof(float) +                 // xs, hs, aggs
      (size_t)(NN * FS + WSW_N) * sizeof(_Float16) +          // ah, wsw
      (size_t)(NN * 2 + H64 * 3 + 4) * sizeof(float);         // dinv, degs, means, zbuf, red, stat

  ring_policy_kernel<<<2048, 256, shmem, stream>>>(
      node_feature, edge_index, emb, gcn_w, gcn_b, lin1_w, lin1_b, lin2_w, lin2_b,
      ln_g, ln_b, ex_w1, ex_b1, ex_ln_g, ex_ln_b, ex_w2, ex_b2, out);
}
